// CausalSelfAttention_13486197310195
// MI455X (gfx1250) — matrix operand order repaired and checked by reading, not compiled, not run on hardware
//
#include <hip/hip_runtime.h>
#include <hip/hip_bf16.h>

#define TT 2048
#define CC 1024
#define HH 16
#define KVH 4
#define DD 64
#define KVC 256
#define NQKV 1536
#define SCALE_F 0.125f      /* 1/sqrt(64) */
#define EPS_F 1e-6f

typedef __attribute__((ext_vector_type(16))) __bf16 v16bf;
typedef __attribute__((ext_vector_type(8)))  float v8f;
typedef __attribute__((ext_vector_type(4)))  unsigned int v4u;

union Frag { v16bf v; v4u q[2]; unsigned int u[8]; unsigned short s[16]; };

__device__ __forceinline__ unsigned short f2bf(float f) {
    unsigned int u = __float_as_uint(f);
    unsigned int r = (u + 0x7fffu + ((u >> 16) & 1u)) >> 16;
    return (unsigned short)r;
}

// ---------------- conversion / packing ----------------
__global__ void k_cvt_x(const float* __restrict__ x, unsigned short* __restrict__ xh, int n) {
    int i = blockIdx.x * blockDim.x + threadIdx.x;
    if (i < n) xh[i] = f2bf(x[i]);
}

// Fragment-major B pack: Bp[(((kc*NT16 + nt)*32 + lane)*8 + v)*2 + p] = W[n][k]
//   n = nt*16 + (lane&15);  k = kc*32 + 2*v + (v>=4 ? 8 : 0) + (lane>=16 ? 8 : 0) + p
// -> each lane's 16 bf16 of a B fragment are contiguous (2 x b128 loads).
__global__ void k_pack_qkv(const float* __restrict__ Wq, const float* __restrict__ Wk,
                           const float* __restrict__ Wv, unsigned short* __restrict__ Bp) {
    int o = blockIdx.x * blockDim.x + threadIdx.x;
    if (o >= CC * NQKV) return;
    const int NT16 = NQKV / 16;
    int blk = o >> 9, within = o & 511;
    int lane = within >> 4, v = (within >> 1) & 7, p = within & 1;
    int nt = blk % NT16, kc = blk / NT16;
    int n = nt * 16 + (lane & 15);
    int k = kc * 32 + 2 * v + ((v >= 4) ? 8 : 0) + ((lane >= 16) ? 8 : 0) + p;
    float w;
    if (n < CC)            w = Wq[(size_t)n * CC + k];
    else if (n < CC + KVC) w = Wk[(size_t)(n - CC) * CC + k];
    else                   w = Wv[(size_t)(n - CC - KVC) * CC + k];
    Bp[o] = f2bf(w);
}

__global__ void k_pack_wo(const float* __restrict__ Wo, unsigned short* __restrict__ Bp) {
    int o = blockIdx.x * blockDim.x + threadIdx.x;
    if (o >= CC * CC) return;
    const int NT16 = CC / 16;
    int blk = o >> 9, within = o & 511;
    int lane = within >> 4, v = (within >> 1) & 7, p = within & 1;
    int nt = blk % NT16, kc = blk / NT16;
    int n = nt * 16 + (lane & 15);
    int k = kc * 32 + 2 * v + ((v >= 4) ? 8 : 0) + ((lane >= 16) ? 8 : 0) + p;
    Bp[o] = f2bf(Wo[(size_t)n * CC + k]);
}

// ---------------- WMMA GEMM ----------------
// C[MxN] = A[MxK](bf16, row-major) * Bp (bf16, fragment-major pack)
// block = 128 (4 waves); each wave computes a 64x32 tile (8 accumulators = 64 VGPRs,
// no spills); grid = (N/32, M/256)
__global__ void __launch_bounds__(128, 1)
k_gemm_bf16(const unsigned short* __restrict__ A, const unsigned short* __restrict__ Bp,
            float* __restrict__ C, int M, int N, int K, int ldc) {
    int wave = threadIdx.x >> 5, lane = threadIdx.x & 31;
    int half = lane >> 4;
    int m = lane & 15, n = lane & 15;
    int m0 = blockIdx.y * 256 + wave * 64;
    int n0 = blockIdx.x * 32;
    const int NT16 = N / 16;
    v8f acc[4][2] = {};
    const unsigned short* arow0 = A + (size_t)(m0 +  0 + m) * K + (half ? 8 : 0);
    const unsigned short* arow1 = A + (size_t)(m0 + 16 + m) * K + (half ? 8 : 0);
    const unsigned short* arow2 = A + (size_t)(m0 + 32 + m) * K + (half ? 8 : 0);
    const unsigned short* arow3 = A + (size_t)(m0 + 48 + m) * K + (half ? 8 : 0);
    const unsigned short* bbase = Bp + ((size_t)(n0 / 16) * 32 + lane) * 16;
    const size_t bkstride = (size_t)NT16 * 512;   // one k-chunk step in packed B

    for (int kc = 0; kc < K / 32; kc++) {
        int k0 = kc * 32;
        Frag a[4];
        a[0].q[0] = *(const v4u*)(arow0 + k0);      a[0].q[1] = *(const v4u*)(arow0 + k0 + 16);
        a[1].q[0] = *(const v4u*)(arow1 + k0);      a[1].q[1] = *(const v4u*)(arow1 + k0 + 16);
        a[2].q[0] = *(const v4u*)(arow2 + k0);      a[2].q[1] = *(const v4u*)(arow2 + k0 + 16);
        a[3].q[0] = *(const v4u*)(arow3 + k0);      a[3].q[1] = *(const v4u*)(arow3 + k0 + 16);
        if (k0 + 64 < K) {
            __builtin_prefetch(arow0 + k0 + 64, 0, 0);
            __builtin_prefetch(arow3 + k0 + 64, 0, 0);
        }
        const unsigned short* bk = bbase + (size_t)kc * bkstride;
#pragma unroll
        for (int nt = 0; nt < 2; nt++) {
            Frag b;
            b.q[0] = *(const v4u*)(bk + nt * 512);
            b.q[1] = *(const v4u*)(bk + nt * 512 + 8);
#pragma unroll
            for (int mi = 0; mi < 4; mi++)
                acc[mi][nt] = __builtin_amdgcn_wmma_f32_16x16x32_bf16(
                    false, a[mi].v, false, b.v, (short)0, acc[mi][nt], false, false);
        }
    }
#pragma unroll
    for (int mi = 0; mi < 4; mi++)
#pragma unroll
        for (int nt = 0; nt < 2; nt++)
#pragma unroll
            for (int r = 0; r < 8; r++)
                C[(size_t)(m0 + mi * 16 + half * 8 + r) * ldc + n0 + nt * 16 + n] = acc[mi][nt][r];
}

// ---------------- rmsnorm + RoPE + head split epilogue ----------------
// QKVraw: [T][1536] f32 (Q | K | V).  Outputs: Qh[H][T][64], Kh[KVH][T][64] bf16, Vt[KVH][64][T] bf16.
__global__ void k_rms_rope(const float* __restrict__ QKV,
                           const float* __restrict__ qw, const float* __restrict__ kw,
                           unsigned short* __restrict__ Qh, unsigned short* __restrict__ Kh,
                           unsigned short* __restrict__ Vt) {
    __shared__ float red[256];
    int t = blockIdx.x, tid = threadIdx.x;
    const float* row = QKV + (size_t)t * NQKV;
    float ss = 0.f;
    for (int c = tid; c < CC; c += 256) { float v = row[c]; ss += v * v; }
    red[tid] = ss; __syncthreads();
    for (int s = 128; s > 0; s >>= 1) { if (tid < s) red[tid] += red[tid + s]; __syncthreads(); }
    float rq = rsqrtf(red[0] / (float)CC + EPS_F);
    __syncthreads();
    float sk = 0.f;
    for (int c = tid; c < KVC; c += 256) { float v = row[CC + c]; sk += v * v; }
    red[tid] = sk; __syncthreads();
    for (int s = 128; s > 0; s >>= 1) { if (tid < s) red[tid] += red[tid + s]; __syncthreads(); }
    float rk = rsqrtf(red[0] / (float)KVC + EPS_F);
    __syncthreads();

    const float LG = 0.28782313662425572f; // ln(10000)/32
    for (int c = tid; c < CC; c += 256) {
        int h = c >> 6, d = c & 63;
        int pc = (d < 32) ? c + 32 : c - 32;
        float qn = row[c] * rq * qw[c];
        float qp = row[pc] * rq * qw[pc];
        int fi = d & 31;
        float fr = (float)t * __expf(-(float)fi * LG);
        float cs = __cosf(fr), sn = __sinf(fr);
        float o = (d < 32) ? (qn * cs - qp * sn) : (qn * cs + qp * sn);
        Qh[(size_t)h * TT * DD + (size_t)t * DD + d] = f2bf(o);
    }
    for (int c = tid; c < KVC; c += 256) {
        int h = c >> 6, d = c & 63;
        int pc = (d < 32) ? c + 32 : c - 32;
        float kn = row[CC + c] * rk * kw[c];
        float kp = row[CC + pc] * rk * kw[pc];
        int fi = d & 31;
        float fr = (float)t * __expf(-(float)fi * LG);
        float cs = __cosf(fr), sn = __sinf(fr);
        float o = (d < 32) ? (kn * cs - kp * sn) : (kn * cs + kp * sn);
        Kh[(size_t)h * TT * DD + (size_t)t * DD + d] = f2bf(o);
        Vt[(size_t)h * DD * TT + (size_t)d * TT + t] = f2bf(row[CC + KVC + c]);
    }
}

// ---------------- FoX forgetting gate ----------------
__global__ void k_gate(const float* __restrict__ x, const float* __restrict__ wl,
                       const float* __restrict__ fw, const float* __restrict__ fb,
                       float* __restrict__ logf_buf) {
    int wid = blockIdx.x * 8 + (threadIdx.x >> 5);
    int lane = threadIdx.x & 31;
    int t = wid >> 4, h = wid & 15;
    float a = 0.f, b = 0.f;
    for (int s = lane; s < CC; s += 32) {
        float xv = x[(size_t)t * CC + s];
        a += xv * wl[(size_t)s * HH + h];
        b += xv * fw[(size_t)h * CC + s];
    }
#pragma unroll
    for (int off = 1; off < 32; off <<= 1) {
        a += __shfl_xor(a, off, 32);
        b += __shfl_xor(b, off, 32);
    }
    if (lane == 0) {
        float lam = ((a > 0.f) ? a : expm1f(a)) + 1.0f;
        float logit = (b + fb[h]) * lam;
        float ls = (logit >= 0.f) ? -log1pf(__expf(-logit)) : (logit - log1pf(__expf(logit)));
        logf_buf[(size_t)h * TT + t] = ls / (lam + 1e-3f);
    }
}

__global__ void k_cumsum(const float* __restrict__ lf, float* __restrict__ cc) {
    int h = blockIdx.x, lane = threadIdx.x;
    float carry = 0.f;
    for (int base = 0; base < TT; base += 32) {
        float v = lf[(size_t)h * TT + base + lane];
#pragma unroll
        for (int off = 1; off < 32; off <<= 1) {
            float u = __shfl_up(v, off, 32);
            if (lane >= off) v += u;
        }
        cc[(size_t)h * TT + base + lane] = v + carry;
        carry += __shfl(v, 31, 32);
    }
}

// ---------------- flash attention with FoX bias ----------------
// grid (H, T/64), block 128 (4 waves, each 16 query rows)
__global__ void __launch_bounds__(128)
k_attn(const unsigned short* __restrict__ Qh, const unsigned short* __restrict__ Kh,
       const unsigned short* __restrict__ Vt, const float* __restrict__ Cc,
       unsigned short* __restrict__ Yh) {
    __shared__ __align__(16) unsigned short Plds[4][512]; // per-wave 16x32 bf16 P tile
    int h = blockIdx.x;
    int wave = threadIdx.x >> 5, lane = threadIdx.x & 31;
    int half = lane >> 4;
    int m = lane & 15, n = lane & 15;
    int t0 = blockIdx.y * 64 + wave * 16;
    int kvh = h >> 2;
    const unsigned short* qp = Qh + (size_t)h * TT * DD;
    const unsigned short* kp = Kh + (size_t)kvh * TT * DD;
    const unsigned short* vp = Vt + (size_t)kvh * DD * TT;
    const float* ch = Cc + (size_t)h * TT;
    unsigned short* pl = Plds[wave];

    // Q A-fragments for d-chunks [0,32) and [32,64): each = 2 x b128
    Frag aq0, aq1;
    const unsigned short* qrow = qp + (size_t)(t0 + m) * DD + (half ? 16 : 0);
    aq0.q[0] = *(const v4u*)(qrow);       aq0.q[1] = *(const v4u*)(qrow + 8);
    aq1.q[0] = *(const v4u*)(qrow + 32);  aq1.q[1] = *(const v4u*)(qrow + 40);

    float ci[8], mrow[8], lrow[8];
#pragma unroll
    for (int r = 0; r < 8; r++) {
        ci[r] = ch[t0 + half * 8 + r];
        mrow[r] = -INFINITY; lrow[r] = 0.f;
    }
    v8f acc0 = {}, acc1 = {}, acc2 = {}, acc3 = {};

    for (int j0 = 0; j0 <= t0 + 15; j0 += 32) {
        // ---- scores: two 16-key subtiles, K(depth)=64 via two wmma each ----
        v8f s0 = {}, s1 = {};
        Frag b;
        const unsigned short* kr0 = kp + (size_t)(j0 + n) * DD + (half ? 16 : 0);
        const unsigned short* kr1 = kp + (size_t)(j0 + 16 + n) * DD + (half ? 16 : 0);
        b.q[0] = *(const v4u*)(kr0);     b.q[1] = *(const v4u*)(kr0 + 8);
        s0 = __builtin_amdgcn_wmma_f32_16x16x32_bf16(false, aq0.v, false, b.v, (short)0, s0, false, false);
        b.q[0] = *(const v4u*)(kr0 + 32); b.q[1] = *(const v4u*)(kr0 + 40);
        s0 = __builtin_amdgcn_wmma_f32_16x16x32_bf16(false, aq1.v, false, b.v, (short)0, s0, false, false);
        b.q[0] = *(const v4u*)(kr1);     b.q[1] = *(const v4u*)(kr1 + 8);
        s1 = __builtin_amdgcn_wmma_f32_16x16x32_bf16(false, aq0.v, false, b.v, (short)0, s1, false, false);
        b.q[0] = *(const v4u*)(kr1 + 32); b.q[1] = *(const v4u*)(kr1 + 40);
        s1 = __builtin_amdgcn_wmma_f32_16x16x32_bf16(false, aq1.v, false, b.v, (short)0, s1, false, false);

        float cj0 = ch[j0 + n], cj1 = ch[j0 + 16 + n];
        // ---- bias + causal mask + online softmax ----
#pragma unroll
        for (int r = 0; r < 8; r++) {
            int i = t0 + half * 8 + r;
            float v0 = s0[r] * SCALE_F + ci[r] - cj0;
            float v1 = s1[r] * SCALE_F + ci[r] - cj1;
            if (j0 + n > i)      v0 = -INFINITY;
            if (j0 + 16 + n > i) v1 = -INFINITY;
            float tm = fmaxf(v0, v1);
#pragma unroll
            for (int off = 1; off < 16; off <<= 1) tm = fmaxf(tm, __shfl_xor(tm, off, 32));
            float mnew = fmaxf(mrow[r], tm);
            float alpha = __expf(mrow[r] - mnew);
            float p0 = __expf(v0 - mnew);
            float p1 = __expf(v1 - mnew);
            float rs = p0 + p1;
#pragma unroll
            for (int off = 1; off < 16; off <<= 1) rs += __shfl_xor(rs, off, 32);
            lrow[r] = lrow[r] * alpha + rs;
            mrow[r] = mnew;
            acc0[r] *= alpha; acc1[r] *= alpha; acc2[r] *= alpha; acc3[r] *= alpha;
            int mm = half * 8 + r;
            pl[mm * 32 + n]      = f2bf(p0);
            pl[mm * 32 + 16 + n] = f2bf(p1);
        }
        // ---- reload P as A-fragment (16x32) via 2 x ds_load_b128, do PV ----
        Frag ap;
        const unsigned short* pr = pl + m * 32 + (half ? 16 : 0);
        ap.q[0] = *(const v4u*)(pr);  ap.q[1] = *(const v4u*)(pr + 8);

        Frag bv;
        const unsigned short* vr0 = vp + (size_t)(0 * 16 + n) * TT + j0 + (half ? 16 : 0);
        const unsigned short* vr1 = vp + (size_t)(1 * 16 + n) * TT + j0 + (half ? 16 : 0);
        const unsigned short* vr2 = vp + (size_t)(2 * 16 + n) * TT + j0 + (half ? 16 : 0);
        const unsigned short* vr3 = vp + (size_t)(3 * 16 + n) * TT + j0 + (half ? 16 : 0);
        bv.q[0] = *(const v4u*)(vr0); bv.q[1] = *(const v4u*)(vr0 + 8);
        acc0 = __builtin_amdgcn_wmma_f32_16x16x32_bf16(false, ap.v, false, bv.v, (short)0, acc0, false, false);
        bv.q[0] = *(const v4u*)(vr1); bv.q[1] = *(const v4u*)(vr1 + 8);
        acc1 = __builtin_amdgcn_wmma_f32_16x16x32_bf16(false, ap.v, false, bv.v, (short)0, acc1, false, false);
        bv.q[0] = *(const v4u*)(vr2); bv.q[1] = *(const v4u*)(vr2 + 8);
        acc2 = __builtin_amdgcn_wmma_f32_16x16x32_bf16(false, ap.v, false, bv.v, (short)0, acc2, false, false);
        bv.q[0] = *(const v4u*)(vr3); bv.q[1] = *(const v4u*)(vr3 + 8);
        acc3 = __builtin_amdgcn_wmma_f32_16x16x32_bf16(false, ap.v, false, bv.v, (short)0, acc3, false, false);
    }
    // ---- normalize + store y (bf16, [T][C], col = h*64 + d) ----
#pragma unroll
    for (int r = 0; r < 8; r++) {
        int i = t0 + half * 8 + r;
        float inv = 1.0f / lrow[r];
        size_t base = (size_t)i * CC + (size_t)h * DD;
        Yh[base + 0 * 16 + n] = f2bf(acc0[r] * inv);
        Yh[base + 1 * 16 + n] = f2bf(acc1[r] * inv);
        Yh[base + 2 * 16 + n] = f2bf(acc2[r] * inv);
        Yh[base + 3 * 16 + n] = f2bf(acc3[r] * inv);
    }
}

// ---------------- launch ----------------
extern "C" void kernel_launch(void* const* d_in, const int* in_sizes, int n_in,
                              void* d_out, int out_size, void* d_ws, size_t ws_size,
                              hipStream_t stream) {
    const float* x  = (const float*)d_in[0];
    const float* Wq = (const float*)d_in[1];
    const float* Wk = (const float*)d_in[2];
    const float* Wv = (const float*)d_in[3];
    const float* Wo = (const float*)d_in[4];
    const float* qw = (const float*)d_in[5];
    const float* kw = (const float*)d_in[6];
    const float* fw = (const float*)d_in[7];
    const float* fb = (const float*)d_in[8];
    const float* wl = (const float*)d_in[9];

    char* ws = (char*)d_ws;
    size_t off = 0;
    unsigned short* xh    = (unsigned short*)(ws + off); off += (size_t)TT * CC * 2;
    unsigned short* wqkvp = (unsigned short*)(ws + off); off += (size_t)CC * NQKV * 2;
    unsigned short* wop   = (unsigned short*)(ws + off); off += (size_t)CC * CC * 2;
    float*          qkvr  = (float*)(ws + off);          off += (size_t)TT * NQKV * 4;
    unsigned short* qh    = (unsigned short*)(ws + off); off += (size_t)HH * TT * DD * 2;
    unsigned short* kh    = (unsigned short*)(ws + off); off += (size_t)KVH * TT * DD * 2;
    unsigned short* vt    = (unsigned short*)(ws + off); off += (size_t)KVH * DD * TT * 2;
    float*          lf    = (float*)(ws + off);          off += (size_t)HH * TT * 4;
    float*          cc    = (float*)(ws + off);          off += (size_t)HH * TT * 4;
    unsigned short* yh    = (unsigned short*)(ws + off); off += (size_t)TT * CC * 2;

    k_cvt_x   <<<(TT * CC) / 256, 256, 0, stream>>>(x, xh, TT * CC);
    k_pack_qkv<<<(CC * NQKV) / 256, 256, 0, stream>>>(Wq, Wk, Wv, wqkvp);
    k_pack_wo <<<(CC * CC) / 256, 256, 0, stream>>>(Wo, wop);

    k_gemm_bf16<<<dim3(NQKV / 32, TT / 256), 128, 0, stream>>>(xh, wqkvp, qkvr, TT, NQKV, CC, NQKV);

    k_rms_rope<<<TT, 256, 0, stream>>>(qkvr, qw, kw, qh, kh, vt);
    k_gate    <<<(TT * HH) / 8, 256, 0, stream>>>(x, wl, fw, fb, lf);
    k_cumsum  <<<HH, 32, 0, stream>>>(lf, cc);

    k_attn<<<dim3(HH, TT / 64), 128, 0, stream>>>(qh, kh, vt, cc, yh);

    k_gemm_bf16<<<dim3(CC / 32, TT / 256), 128, 0, stream>>>(yh, wop, (float*)d_out, TT, CC, CC, CC);
}
